// GATEncoder_75771813036523
// MI455X (gfx1250) — matrix operand order repaired and checked by reading, not compiled, not run on hardware
//
#include <hip/hip_runtime.h>
#include <math.h>

// ---------------- problem constants (from reference) ----------------
#define N_NODES 20000
#define E_EDGES 320000
#define ETOT    (E_EDGES + N_NODES)
#define F_IN    64
#define HIDDIM  64
#define NHEADS  4
#define HCDIM   256
#define NGROUP  64
#define NHID_   256
#define NOUT_   128
#define INV_SQRT2F 0.70710678118654752440f
#define MPAD_MAX 20096   // ceil(N_NODES/128)*128

typedef __attribute__((ext_vector_type(16))) __bf16 v16bf;
typedef __attribute__((ext_vector_type(8)))  __bf16 v8bf;
typedef __attribute__((ext_vector_type(8)))  float  v8f;
typedef __attribute__((ext_vector_type(4)))  unsigned int u32x4;
typedef __attribute__((ext_vector_type(8)))  int i32x8;
typedef __attribute__((ext_vector_type(4)))  int i32x4;

// Tensor Data Mover availability (device pass only; host pass falls back)
#if defined(__has_builtin)
#if __has_builtin(__builtin_amdgcn_tensor_load_to_lds) && \
    __has_builtin(__builtin_amdgcn_s_wait_tensorcnt)
#define HAVE_TDM 1
#else
#define HAVE_TDM 0
#endif
#else
#define HAVE_TDM 0
#endif

// ============================================================================
// fp32 -> bf16 pre-pack (zero-pads rows up to the 128-row GEMM tile so the
// WMMA loop needs no bounds predication on A loads)
// ============================================================================
__global__ void pack_bf16_kernel(const float* __restrict__ a,
                                 __bf16* __restrict__ o, int n, int npad) {
  const int i = blockIdx.x * blockDim.x + threadIdx.x;
  if (i < n)          o[i] = (__bf16)a[i];
  else if (i < npad)  o[i] = (__bf16)0.0f;
}

// W[K,Ncol] fp32 -> Wt[Ncol,K] bf16 (transpose so a B column is contiguous)
__global__ void pack_wt_kernel(const float* __restrict__ w, __bf16* __restrict__ wt,
                               int K, int Ncol) {
  const int i = blockIdx.x * blockDim.x + threadIdx.x;
  if (i >= K * Ncol) return;
  const int k = i / Ncol, n = i - k * Ncol;
  wt[(size_t)n * K + k] = (__bf16)w[i];   // read coalesced, write 2B scatter (tiny)
}

// ============================================================================
// WMMA GEMM on pre-packed bf16: C[M,Ncol] = A[Mpad,K] @ Bt[Ncol,K]^T
// Block = 256 threads = 8 waves; tile 128 rows x 32 cols.  The 32-column B
// slab (32*K bf16, <=16KB, contiguous in Bt) is staged once per block into
// LDS via the Tensor Data Mover; all waves read B fragments with ds_load.
// A streams from global unconditionally (rows zero-padded to the tile) with
// b128 loads + speculative global_prefetch_b8.  f32 acc, wmma 16x16x32 bf16.
// ============================================================================
#define LDSB_ELEMS (32 * 256)

__global__ __launch_bounds__(256)
void gemm_bf16_wmma_kernel(const __bf16* __restrict__ A, const __bf16* __restrict__ Bt,
                           const float* __restrict__ bias, float* __restrict__ C,
                           int M, int K, int Ncol, int flags) {
  __shared__ __bf16 ldsB[LDSB_ELEMS];
  const int colBase = blockIdx.y * 32;
  const __bf16* slab = Bt + (size_t)colBase * K;   // 32*K contiguous bf16

#if HAVE_TDM
  if (threadIdx.x < 32) {                          // one wave issues the DMA
    const unsigned T = (unsigned)(32 * K);         // slab length in elements
    const unsigned long long ga = (unsigned long long)(uintptr_t)slab;
    const unsigned laddr = (unsigned)(uintptr_t)(&ldsB[0]);
    u32x4 g0;
    g0[0] = 1u;                                        // count=1 (valid), user D#
    g0[1] = laddr;                                     // lds_addr (bytes)
    g0[2] = (unsigned)(ga & 0xFFFFFFFFull);            // global_addr[31:0]
    g0[3] = (unsigned)((ga >> 32) & 0x1FFFFFFull)      // global_addr[56:32]
            | (2u << 30);                              // type = 2 ("image")
    i32x8 g1;
    g1[0] = (int)0x00010000u;                          // data_size = 1 -> 2 bytes
    g1[1] = (int)((T & 0xFFFFu) << 16);                // tensor_dim0[15:0]
    g1[2] = (int)((T >> 16) | (1u << 16));             // tensor_dim0[31:16] | dim1=1
    g1[3] = (int)((T & 0xFFFFu) << 16);                // tile_dim0 = T (<= 8192)
    g1[4] = 1;                                         // tile_dim1 = 1
    g1[5] = (int)T;                                    // tensor_dim0_stride lo
    g1[6] = 0; g1[7] = 0;
    i32x4 gz = {0, 0, 0, 0};
#if __clang_major__ >= 23
    i32x8 gz8 = {0, 0, 0, 0, 0, 0, 0, 0};
    __builtin_amdgcn_tensor_load_to_lds(g0, g1, gz, gz, gz8, 0);
#else
    __builtin_amdgcn_tensor_load_to_lds(g0, g1, gz, gz, 0);
#endif
    __builtin_amdgcn_s_wait_tensorcnt(0);
  }
  __syncthreads();
#else
  {
    const int vecs = (32 * K) / 8;                 // cooperative v8bf copy
    for (int idx = threadIdx.x; idx < vecs; idx += 256)
      ((v8bf*)ldsB)[idx] = ((const v8bf*)slab)[idx];
    __syncthreads();
  }
#endif

  const int wave = threadIdx.x >> 5;
  const int lane = threadIdx.x & 31;
  const int half = lane >> 4;
  const int l16  = lane & 15;
  const int rowBase = blockIdx.x * 128 + wave * 16;

  // A rows are zero-padded up to gridDim.x*128, so loads need no predication.
  const __bf16* arow = A + (size_t)(rowBase + l16) * K;
  const __bf16* bl0 = ldsB + (size_t)l16 * K;               // column colBase+l16
  const __bf16* bl1 = ldsB + (size_t)(l16 + 16) * K;

  v8f acc0 = {}; v8f acc1 = {};
  for (int k0 = 0; k0 < K; k0 += 32) {
    // speculative prefetch of next A K-slice; past-the-end is silently dropped
    __builtin_prefetch(arow + k0 + 32 + half * 8, 0, 1);
    // A fragment: 16 contiguous elements [half*16, half*16+16)
    const v8bf alo = *(const v8bf*)(arow + k0 + half * 16);
    const v8bf ahi = *(const v8bf*)(arow + k0 + half * 16 + 8);
    const v16bf a = __builtin_shufflevector(alo, ahi, 0, 1, 2, 3, 4, 5, 6, 7,
                                            8, 9, 10, 11, 12, 13, 14, 15);
    // B fragments from LDS: 16 contiguous bf16 at k0 + half*16 within a column
    const v16bf b0 = *(const v16bf*)(bl0 + k0 + half * 16);
    const v16bf b1 = *(const v16bf*)(bl1 + k0 + half * 16);
    acc0 = __builtin_amdgcn_wmma_f32_16x16x32_bf16(false, a, false, b0,
                                                   (short)0, acc0, false, false);
    acc1 = __builtin_amdgcn_wmma_f32_16x16x32_bf16(false, a, false, b1,
                                                   (short)0, acc1, false, false);
  }
  // C/D layout: VGPR rv -> M = rv + 8*half, N = lane%16 within tile
#pragma unroll
  for (int rv = 0; rv < 8; ++rv) {
    const int row = rowBase + rv + 8 * half;
    if (row < M) {
      float v0 = acc0[rv], v1 = acc1[rv];
      const int c0 = colBase + l16, c1 = colBase + 16 + l16;
      if (flags & 1) { v0 += bias[c0]; v1 += bias[c1]; }
      if (flags & 2) { v0 = fmaxf(v0, 0.0f); v1 = fmaxf(v1, 0.0f); }
      C[(size_t)row * Ncol + c0] = v0;
      C[(size_t)row * Ncol + c1] = v1;
    }
  }
}

// ============================================================================
// Attention logits: al_s[n,h] = sum_c h[n,h,c]*a_s[h,c] (a_* flat == [HCDIM])
// ============================================================================
__global__ __launch_bounds__(256)
void attn_logits_kernel(const float* __restrict__ h, const float* __restrict__ a_s,
                        const float* __restrict__ a_d, float* __restrict__ als,
                        float* __restrict__ ald, int heads) {
  __shared__ float s1[HCDIM], s2[HCDIM];
  const int n = blockIdx.x;
  const int c = threadIdx.x;
  const float hv = h[(size_t)n * HCDIM + c];
  s1[c] = hv * a_s[c];
  s2[c] = hv * a_d[c];
  __syncthreads();
  const int C = HCDIM / heads;  // 64 or 256, power of two
  for (int off = C >> 1; off > 0; off >>= 1) {
    if ((c & (C - 1)) < off) { s1[c] += s1[c + off]; s2[c] += s2[c + off]; }
    __syncthreads();
  }
  if ((c & (C - 1)) == 0) {
    const int hh = c / C;
    als[n * heads + hh] = s1[c];
    ald[n * heads + hh] = s2[c];
  }
}

// ---------------- edge helpers ----------------
static __device__ __forceinline__ void edge_sd(const int* __restrict__ ei,
                                               int e, int& s, int& d) {
  if (e < E_EDGES) { s = ei[e]; d = ei[E_EDGES + e]; }
  else             { s = e - E_EDGES; d = s; }          // self-loops
}

static __device__ __forceinline__ void atomicMaxFloat(float* addr, float val) {
  if (val >= 0.0f) atomicMax((int*)addr, __float_as_int(val));
  else             atomicMin((unsigned int*)addr, __float_as_uint(val));
}

__global__ void edge_max_kernel(const int* __restrict__ ei,
                                const float* __restrict__ als,
                                const float* __restrict__ ald,
                                float* __restrict__ mmax, int heads) {
  const int i = blockIdx.x * blockDim.x + threadIdx.x;
  if (i >= ETOT * heads) return;
  const int e = i / heads, hh = i - e * heads;
  int s, d; edge_sd(ei, e, s, d);
  float v = als[s * heads + hh] + ald[d * heads + hh];
  v = (v > 0.0f) ? v : 0.2f * v;                         // leaky_relu(0.2)
  atomicMaxFloat(&mmax[d * heads + hh], v);
}

__global__ void edge_expsum_kernel(const int* __restrict__ ei,
                                   const float* __restrict__ als,
                                   const float* __restrict__ ald,
                                   const float* __restrict__ mmax,
                                   float* __restrict__ exbuf,
                                   float* __restrict__ den, int heads) {
  const int i = blockIdx.x * blockDim.x + threadIdx.x;
  if (i >= ETOT * heads) return;
  const int e = i / heads, hh = i - e * heads;
  int s, d; edge_sd(ei, e, s, d);
  float v = als[s * heads + hh] + ald[d * heads + hh];
  v = (v > 0.0f) ? v : 0.2f * v;
  const float xv = expf(v - mmax[d * heads + hh]);
  exbuf[i] = xv;
  atomicAdd(&den[d * heads + hh], xv);
}

// One 256-thread block per edge: out[dst, c] += h[src, c] * coef[head(c)]
__global__ __launch_bounds__(256)
void edge_agg_kernel(const int* __restrict__ ei, const float* __restrict__ exbuf,
                     const float* __restrict__ den, const float* __restrict__ hsrc,
                     float* __restrict__ out, int heads) {
  const int e = blockIdx.x;
  const int c = threadIdx.x;
  int s, d; edge_sd(ei, e, s, d);
  const int C = HCDIM / heads;
  const int hh = c / C;
  const float coef = exbuf[e * heads + hh] / den[d * heads + hh];
  atomicAdd(&out[(size_t)d * HCDIM + c], hsrc[(size_t)s * HCDIM + c] * coef);
}

// ---------------- BatchNorm (training stats) + ReLU ----------------
__global__ __launch_bounds__(256)
void bn_stats_kernel(const float* __restrict__ x, float* __restrict__ sum,
                     float* __restrict__ sumsq) {
  const int c  = threadIdx.x;            // channel, blockDim == HCDIM
  const int r0 = blockIdx.x * 128;
  const int r1 = (r0 + 128 < N_NODES) ? (r0 + 128) : N_NODES;
  float s = 0.0f, sq = 0.0f;
  for (int r = r0; r < r1; ++r) {
    const float v = x[(size_t)r * HCDIM + c];
    s += v; sq += v * v;
  }
  atomicAdd(&sum[c], s);
  atomicAdd(&sumsq[c], sq);
}

__global__ void bn_apply_kernel(float* __restrict__ x, const float* __restrict__ sum,
                                const float* __restrict__ sumsq,
                                const float* __restrict__ g,
                                const float* __restrict__ be) {
  const int i = blockIdx.x * blockDim.x + threadIdx.x;
  if (i >= N_NODES * HCDIM) return;
  const int c = i & (HCDIM - 1);
  const float inv_n = 1.0f / (float)N_NODES;
  const float mu  = sum[c] * inv_n;
  const float var = sumsq[c] * inv_n - mu * mu;
  const float v = (x[i] - mu) * rsqrtf(var + 1e-5f);
  x[i] = fmaxf(g[c] * v + be[c], 0.0f);
}

// ---------------- misc elementwise ----------------
__global__ void fill_kernel(float* __restrict__ p, float v, int n) {
  const int i = blockIdx.x * blockDim.x + threadIdx.x;
  if (i < n) p[i] = v;
}

__global__ void bias_rows_kernel(float* __restrict__ p, const float* __restrict__ b,
                                 int n, int cols) {
  const int i = blockIdx.x * blockDim.x + threadIdx.x;
  if (i < n) p[i] = b[i % cols];
}

__global__ void addres_kernel(float* __restrict__ h, const float* __restrict__ r) {
  const int i = blockIdx.x * blockDim.x + threadIdx.x;
  if (i < N_NODES * HCDIM) h[i] = (h[i] + r[i]) * INV_SQRT2F;
}

__global__ void pool_kernel(const float* __restrict__ x, const int* __restrict__ batch,
                            float* __restrict__ pool, float* __restrict__ cnt) {
  const int i = blockIdx.x * blockDim.x + threadIdx.x;
  if (i >= N_NODES * HIDDIM) return;
  const int n = i / HIDDIM, c = i - n * HIDDIM;
  const int g = batch[n];
  atomicAdd(&pool[g * HIDDIM + c], x[i]);
  if (c == 0) atomicAdd(&cnt[g], 1.0f);
}

__global__ void pool_div_kernel(float* __restrict__ pool, const float* __restrict__ cnt) {
  const int i = blockIdx.x * blockDim.x + threadIdx.x;
  if (i >= NGROUP * HIDDIM) return;
  pool[i] /= fmaxf(cnt[i / HIDDIM], 1.0f);
}

// ============================================================================
// Host-side orchestration
// ============================================================================
struct WsPtrs {
  float *h_lin, *agg, *res, *als, *ald, *mmax, *den, *exbuf, *stats;
  __bf16 *abf, *wtb;   // bf16 staging for GEMM operands
};

static void gemm(hipStream_t s, const float* A, const float* W, const float* bias,
                 float* C, int M, int K, int Ncol, int flags, const WsPtrs& w) {
  const int Mpad = ((M + 127) / 128) * 128;
  const int na = M * K, npad = Mpad * K, nw = K * Ncol;
  pack_bf16_kernel<<<(npad + 255) / 256, 256, 0, s>>>(A, w.abf, na, npad);
  pack_wt_kernel<<<(nw + 255) / 256, 256, 0, s>>>(W, w.wtb, K, Ncol);
  dim3 grid(Mpad / 128, Ncol / 32);
  gemm_bf16_wmma_kernel<<<grid, 256, 0, s>>>(w.abf, w.wtb, bias, C, M, K, Ncol, flags);
}

static void gat_layer(hipStream_t s, const float* xin, int cin, int heads,
                      const float* W, const float* a_s, const float* a_d,
                      const float* bias, const int* ei, const WsPtrs& w) {
  // h = xin @ W : [N, 256]
  gemm(s, xin, W, nullptr, w.h_lin, N_NODES, cin, HCDIM, 0, w);
  attn_logits_kernel<<<N_NODES, HCDIM, 0, s>>>(w.h_lin, a_s, a_d, w.als, w.ald, heads);
  const int nh = N_NODES * heads;
  fill_kernel<<<(nh + 255) / 256, 256, 0, s>>>(w.mmax, -INFINITY, nh);
  fill_kernel<<<(nh + 255) / 256, 256, 0, s>>>(w.den, 0.0f, nh);
  const int tot = ETOT * heads;
  edge_max_kernel<<<(tot + 255) / 256, 256, 0, s>>>(ei, w.als, w.ald, w.mmax, heads);
  edge_expsum_kernel<<<(tot + 255) / 256, 256, 0, s>>>(ei, w.als, w.ald, w.mmax,
                                                       w.exbuf, w.den, heads);
  // agg initialized with layer bias (reference adds bias after segment-sum)
  bias_rows_kernel<<<(N_NODES * HCDIM + 255) / 256, 256, 0, s>>>(w.agg, bias,
                                                                 N_NODES * HCDIM, HCDIM);
  edge_agg_kernel<<<ETOT, HCDIM, 0, s>>>(ei, w.exbuf, w.den, w.h_lin, w.agg, heads);
}

static void bn_relu(hipStream_t s, float* xbuf, const float* g, const float* be,
                    const WsPtrs& w) {
  fill_kernel<<<(2 * HCDIM + 255) / 256, 256, 0, s>>>(w.stats, 0.0f, 2 * HCDIM);
  bn_stats_kernel<<<(N_NODES + 127) / 128, HCDIM, 0, s>>>(xbuf, w.stats, w.stats + HCDIM);
  bn_apply_kernel<<<(N_NODES * HCDIM + 255) / 256, 256, 0, s>>>(xbuf, w.stats,
                                                                w.stats + HCDIM, g, be);
}

static void run_block(hipStream_t s, const float* xin, int cin, float* xout,
                      const int* ei, const WsPtrs& w, const float* const* P) {
  // P: W1, a1s, a1d, b1, g1, be1, W2, a2s, a2d, b2, g2, be2, Wr, br, Wf, bf
  gat_layer(s, xin, cin, NHEADS, P[0], P[1], P[2], P[3], ei, w);   // -> w.agg
  bn_relu(s, w.agg, P[4], P[5], w);
  gat_layer(s, w.agg, HCDIM, 1, P[6], P[7], P[8], P[9], ei, w);    // -> w.agg
  bn_relu(s, w.agg, P[10], P[11], w);
  gemm(s, xin, P[12], P[13], w.res, N_NODES, cin, HCDIM, 1, w);    // residual
  addres_kernel<<<(N_NODES * HCDIM + 255) / 256, 256, 0, s>>>(w.agg, w.res);
  gemm(s, w.agg, P[14], P[15], xout, N_NODES, HCDIM, HIDDIM, 1, w); // final proj
}

extern "C" void kernel_launch(void* const* d_in, const int* in_sizes, int n_in,
                              void* d_out, int out_size, void* d_ws, size_t ws_size,
                              hipStream_t stream) {
  (void)in_sizes; (void)n_in; (void)out_size; (void)ws_size;
  const float* x     = (const float*)d_in[0];
  const int*   ei    = (const int*)d_in[1];
  const int*   batch = (const int*)d_in[2];
  const float* P0[16]; const float* P1[16];
  for (int i = 0; i < 16; ++i) {
    P0[i] = (const float*)d_in[3 + i];
    P1[i] = (const float*)d_in[19 + i];
  }
  const float* Wh1 = (const float*)d_in[35];
  const float* bh1 = (const float*)d_in[36];
  const float* Wh2 = (const float*)d_in[37];
  const float* bh2 = (const float*)d_in[38];

  // -------- workspace carve-out (floats); total ~90 MB --------
  float* p = (float*)d_ws;
  WsPtrs w;
  w.h_lin = p; p += (size_t)N_NODES * HCDIM;
  w.agg   = p; p += (size_t)N_NODES * HCDIM;
  w.res   = p; p += (size_t)N_NODES * HCDIM;
  float* xo0 = p; p += (size_t)N_NODES * HIDDIM;
  float* xo1 = p; p += (size_t)N_NODES * HIDDIM;
  w.als   = p; p += (size_t)N_NODES * NHEADS;
  w.ald   = p; p += (size_t)N_NODES * NHEADS;
  w.mmax  = p; p += (size_t)N_NODES * NHEADS;
  w.den   = p; p += (size_t)N_NODES * NHEADS;
  w.exbuf = p; p += (size_t)ETOT * NHEADS;
  w.stats = p; p += 2 * HCDIM;
  float* pool = p; p += NGROUP * HIDDIM;
  float* cnt  = p; p += NGROUP;
  float* mlp1 = p; p += NGROUP * NHID_;
  // bf16 staging (reuse float-aligned tail; sizes in bf16 elements)
  w.abf = (__bf16*)p; p += ((size_t)MPAD_MAX * HCDIM) / 2 + 16;  // Mpad*K bf16
  w.wtb = (__bf16*)p; p += ((size_t)HCDIM * HCDIM) / 2 + 16;     // max K*Ncol bf16

  // -------- two GAT blocks --------
  run_block(stream, x,   F_IN,   xo0, ei, w, P0);
  run_block(stream, xo0, HIDDIM, xo1, ei, w, P1);

  // -------- group mean pool --------
  fill_kernel<<<(NGROUP * HIDDIM + 255) / 256, 256, 0, stream>>>(pool, 0.0f,
                                                                 NGROUP * HIDDIM);
  fill_kernel<<<1, 256, 0, stream>>>(cnt, 0.0f, NGROUP);
  pool_kernel<<<(N_NODES * HIDDIM + 255) / 256, 256, 0, stream>>>(xo1, batch, pool, cnt);
  pool_div_kernel<<<(NGROUP * HIDDIM + 255) / 256, 256, 0, stream>>>(pool, cnt);

  // -------- MLP head --------
  gemm(stream, pool, Wh1, bh1, mlp1, NGROUP, HIDDIM, NHID_, /*bias|relu*/ 3, w);
  gemm(stream, mlp1, Wh2, bh2, (float*)d_out, NGROUP, NHID_, NOUT_, /*bias*/ 1, w);
}
